// MultiheadedAttention_19791209300678
// MI455X (gfx1250) — matrix operand order repaired and checked by reading, not compiled, not run on hardware
//
#include <hip/hip_runtime.h>

// ---------------- problem constants ----------------
constexpr int Bn = 16, Ln = 512, Dn = 512, Hn = 8, DKn = 64;

// ---------------- WMMA types ----------------
typedef __attribute__((ext_vector_type(16))) __bf16          v16bf;
typedef __attribute__((ext_vector_type(16))) unsigned short  v16us;
typedef __attribute__((ext_vector_type(8)))  float           v8f;

union Frag {
    v16us          us;
    v16bf          bf;
    uint4          q[2];
    unsigned short s[16];
};

// native f32 -> bf16 (lets the backend use v_cvt hardware conversion)
static __device__ __forceinline__ unsigned short f2bf(float f) {
    union { __bf16 h; unsigned short u; } cv;
    cv.h = (__bf16)f;
    return cv.u;
}

static __device__ __forceinline__ v8f wmma_bf16(const Frag& a, const Frag& b, v8f c) {
    // (neg_a, A, neg_b, B, c_mod, C, reuse_a, reuse_b)
    return __builtin_amdgcn_wmma_f32_16x16x32_bf16(false, a.bf, false, b.bf,
                                                   (short)0, c, false, false);
}

// A fragment (16x32, M x K) from a row-major f32 matrix; convert to bf16.
// lane 0-15: M=lane, K = 0..7 (v0-3) and 16..23 (v4-7); lane 16-31: +8 on K.
static __device__ __forceinline__ void loadA_f32(Frag& a, const float* src, int ld,
                                                 int row0, int k0, int lane) {
    int r  = row0 + (lane & 15);
    int kb = k0 + ((lane >> 4) << 3);
    const float* p = src + (size_t)r * ld + kb;
    float4 x0 = *(const float4*)(p);
    float4 x1 = *(const float4*)(p + 4);
    float4 x2 = *(const float4*)(p + 16);
    float4 x3 = *(const float4*)(p + 20);
    float t[16] = { x0.x,x0.y,x0.z,x0.w, x1.x,x1.y,x1.z,x1.w,
                    x2.x,x2.y,x2.z,x2.w, x3.x,x3.y,x3.z,x3.w };
#pragma unroll
    for (int e = 0; e < 16; ++e) a.s[e] = f2bf(t[e]);
}

// A fragment from a row-major bf16 matrix (two contiguous 16B loads).
static __device__ __forceinline__ void loadA_bf16(Frag& a, const unsigned short* src,
                                                  int ld, int row0, int k0, int lane) {
    int r  = row0 + (lane & 15);
    int kb = k0 + ((lane >> 4) << 3);
    const unsigned short* p = src + (size_t)r * ld + kb;
    a.q[0] = *(const uint4*)(p);        // K = kb .. kb+7
    a.q[1] = *(const uint4*)(p + 16);   // K = kb+16 .. kb+23
}

// B fragment (32x16, K x N) where B[k][n] = src[n][k] (src row-major [N][K]).
// lane 0-15: N=lane, K = 0..7 and 16..23; lane 16-31: +8 on K.
// Two contiguous 16B loads per lane — the layout every B operand uses below.
static __device__ __forceinline__ void loadB_t_bf16(Frag& b, const unsigned short* src,
                                                    int ld, int n0, int k0, int lane) {
    int n  = n0 + (lane & 15);
    int kb = k0 + ((lane >> 4) << 3);
    const unsigned short* p = src + (size_t)n * ld + kb;
    b.q[0] = *(const uint4*)(p);
    b.q[1] = *(const uint4*)(p + 16);
}

// ---------------- kernel 0: transpose + bf16-convert a [K][N] f32 weight ----
// Wt[n][k] = bf16(W[k][n]); classic 32x32 LDS tile transpose, coalesced both ways.
__global__ __launch_bounds__(256) void k_cvtw(const float* __restrict__ W,
                                              unsigned short* __restrict__ Wt) {
    __shared__ float t[32][33];
    int n0 = blockIdx.x * 32, k0 = blockIdx.y * 32;
    int tx = threadIdx.x & 31, ty = threadIdx.x >> 5;
#pragma unroll
    for (int i = 0; i < 4; ++i)
        t[ty + i * 8][tx] = W[(size_t)(k0 + ty + i * 8) * Dn + n0 + tx];
    __syncthreads();
#pragma unroll
    for (int i = 0; i < 4; ++i)
        Wt[(size_t)(n0 + ty + i * 8) * Dn + k0 + tx] = f2bf(t[tx][ty + i * 8]);
}

// ---------------- kernel 1: projection + head split ----------------
// VT=false: dst[(b*H+h)*L + l][dk]           (row-major per head, for Q/K)
// VT=true : dst[(b*H+h)*DK + dk][L] = x^T    (dk-major, so P@V B-frags are contiguous)
template <bool VT>
__global__ __launch_bounds__(128) void k_proj(const float* __restrict__ X,
                                              const unsigned short* __restrict__ Wt,
                                              unsigned short* __restrict__ dst) {
    int b = blockIdx.z, h = blockIdx.y, l0 = blockIdx.x * 16;
    int tid = threadIdx.x, wave = tid >> 5, lane = tid & 31;
    int lh8 = (lane >> 4) * 8, n15 = lane & 15;
    const float* Xb = X + (size_t)b * Ln * Dn;
    int n0 = h * DKn + wave * 16;
    v8f c = {};
    for (int k0 = 0; k0 < Dn; k0 += 32) {
        if (k0 + 64 < Dn)   // cover L2->WGP latency on the streaming A operand
            __builtin_prefetch(Xb + (size_t)(l0 + (lane & 15)) * Dn + k0 + 64, 0, 0);
        Frag a, bm;
        loadA_f32(a, Xb, Dn, l0, k0, lane);
        loadB_t_bf16(bm, Wt, Dn, n0, k0, lane);
        c = wmma_bf16(a, bm, c);
    }
    size_t base = (size_t)(b * Hn + h) * Ln * DKn;
#pragma unroll
    for (int r = 0; r < 8; ++r) {
        int l = l0 + lh8 + r, dk = wave * 16 + n15;
        if (VT) dst[base + (size_t)dk * Ln + l] = f2bf(c[r]);
        else    dst[base + (size_t)l * DKn + dk] = f2bf(c[r]);
    }
}

// ---------------- kernel 2: attention core ----------------
// per (b,h,16-row q tile): scores -> mask -> softmax -> *aw -> p_attn out -> P@V
__global__ __launch_bounds__(128) void k_attn(const unsigned short* __restrict__ qws,
                                              const unsigned short* __restrict__ kws,
                                              const unsigned short* __restrict__ vT,
                                              const unsigned char*  __restrict__ mask,
                                              const float*          __restrict__ aw,
                                              float*                __restrict__ p_out,
                                              unsigned short*       __restrict__ xh) {
    __shared__ float sc[16 * 512];   // 32 KB score tile (WGP has 320 KB)
    int b = blockIdx.z, h = blockIdx.y, l0 = blockIdx.x * 16;
    int tid = threadIdx.x, wave = tid >> 5, lane = tid & 31;
    int lh8 = (lane >> 4) * 8, n15 = lane & 15;
    size_t bh = (size_t)(b * Hn + h);
    const unsigned short* q  = qws + bh * Ln * DKn;
    const unsigned short* kk = kws + bh * Ln * DKn;
    const unsigned short* v  = vT  + bh * Ln * DKn;   // [DK][L] layout
    const unsigned char*  mk = mask + (size_t)b * Ln * Ln;

    // phase 1: S = (Q Kt) * 1/sqrt(DK), masked -> LDS
    for (int kt = wave; kt < Ln / 16; kt += 4) {
        int n0 = kt * 16;
        v8f c = {};
#pragma unroll
        for (int k0 = 0; k0 < DKn; k0 += 32) {
            Frag a, bm;
            loadA_bf16(a, q, DKn, l0, k0, lane);
            loadB_t_bf16(bm, kk, DKn, n0, k0, lane);   // B[k][n] = K[n][k]
            c = wmma_bf16(a, bm, c);
        }
        int keyc = n0 + n15;
#pragma unroll
        for (int r = 0; r < 8; ++r) {
            int qr = l0 + lh8 + r;
            float s = c[r] * 0.125f;                       // 1/sqrt(64)
            if (mk[(size_t)qr * Ln + keyc]) s = -__builtin_inff();
            sc[(lh8 + r) * 512 + keyc] = s;
        }
    }
    __syncthreads();

    // phase 2: softmax per row, multiply addition weights, emit p_attn
    {
        int row = tid >> 3, sub = tid & 7;   // 16 rows x 8 threads
        float* srow = &sc[row * 512];
        float m = -__builtin_inff();
        for (int j = sub; j < Ln; j += 8) m = fmaxf(m, srow[j]);
#pragma unroll
        for (int o = 1; o < 8; o <<= 1) m = fmaxf(m, __shfl_xor(m, o));
        float s = 0.f;
        for (int j = sub; j < Ln; j += 8) { float e = __expf(srow[j] - m); srow[j] = e; s += e; }
#pragma unroll
        for (int o = 1; o < 8; o <<= 1) s += __shfl_xor(s, o);
        float inv = 1.f / s;
        int qr = l0 + row;
        const float* awr = aw + (size_t)b * Ln * Ln + (size_t)qr * Ln;
        float* pr = p_out + (bh * Ln + qr) * Ln;
        for (int j = sub; j < Ln; j += 8) {
            float p = srow[j] * inv * awr[j];
            srow[j] = p;
            pr[j]   = p;
        }
    }
    __syncthreads();

    // phase 3: X = P @ V  (A from LDS f32->bf16, B contiguous from v^T)
    {
        int n0 = wave * 16;
        v8f c = {};
        for (int k0 = 0; k0 < Ln; k0 += 32) {
            Frag a, bm;
            loadA_f32(a, sc, 512, 0, k0, lane);
            loadB_t_bf16(bm, v, Ln, n0, k0, lane);     // B[k][n] = vT[n][k]
            c = wmma_bf16(a, bm, c);
        }
#pragma unroll
        for (int r = 0; r < 8; ++r) {
            int l = l0 + lh8 + r;
            xh[((size_t)b * Ln + l) * Dn + h * DKn + n0 + n15] = f2bf(c[r]);
        }
    }
}

// ---------------- kernel 3: output projection ----------------
__global__ __launch_bounds__(128) void k_fc(const unsigned short* __restrict__ xh,
                                            const unsigned short* __restrict__ Wt,
                                            float* __restrict__ out) {
    int row0 = blockIdx.x * 16;
    int tid = threadIdx.x, wave = tid >> 5, lane = tid & 31;
    int lh8 = (lane >> 4) * 8, n15 = lane & 15;
    int n0 = blockIdx.y * 64 + wave * 16;
    v8f c = {};
    for (int k0 = 0; k0 < Dn; k0 += 32) {
        if (k0 + 64 < Dn)
            __builtin_prefetch(xh + (size_t)(row0 + (lane & 15)) * Dn + k0 + 64, 0, 0);
        Frag a, bm;
        loadA_bf16(a, xh, Dn, row0, k0, lane);
        loadB_t_bf16(bm, Wt, Dn, n0, k0, lane);
        c = wmma_bf16(a, bm, c);
    }
#pragma unroll
    for (int r = 0; r < 8; ++r)
        out[(size_t)(row0 + lh8 + r) * Dn + n0 + n15] = c[r];
}

// ---------------- host side ----------------
extern "C" void kernel_launch(void* const* d_in, const int* in_sizes, int n_in,
                              void* d_out, int out_size, void* d_ws, size_t ws_size,
                              hipStream_t stream) {
    (void)in_sizes; (void)n_in; (void)out_size; (void)ws_size;
    const float*         query = (const float*)d_in[0];
    const float*         key   = (const float*)d_in[1];
    const float*         value = (const float*)d_in[2];
    const unsigned char* mask  = (const unsigned char*)d_in[3];   // bool [B,1,L,L]
    const float*         aw    = (const float*)d_in[4];
    const float*         wQ    = (const float*)d_in[5];
    const float*         wK    = (const float*)d_in[6];
    const float*         wV    = (const float*)d_in[7];
    const float*         wfc   = (const float*)d_in[8];

    float* out_x = (float*)d_out;                                  // [B,L,D]
    float* out_p = out_x + (size_t)Bn * Ln * Dn;                   // [B,H,L,L]

    size_t M = (size_t)Bn * Hn * Ln * DKn;                         // 4M elems
    size_t W = (size_t)Dn * Dn;                                    // 256K elems
    unsigned short* qws = (unsigned short*)d_ws;
    unsigned short* kws = qws + M;
    unsigned short* vTw = kws + M;                                 // [bh][DK][L]
    unsigned short* xh  = vTw + M;                                 // [B,L,D] bf16
    unsigned short* wQt = xh + M;                                  // [out][in] bf16
    unsigned short* wKt = wQt + W;
    unsigned short* wVt = wKt + W;
    unsigned short* wFt = wVt + W;

    // one-time weight transposes (bf16, [out][in]) -> contiguous B-fragments
    dim3 gt(Dn / 32, Dn / 32), bt(256);
    k_cvtw<<<gt, bt, 0, stream>>>(wQ,  wQt);
    k_cvtw<<<gt, bt, 0, stream>>>(wK,  wKt);
    k_cvtw<<<gt, bt, 0, stream>>>(wV,  wVt);
    k_cvtw<<<gt, bt, 0, stream>>>(wfc, wFt);

    dim3 gp(Ln / 16, Hn, Bn), blk(128);
    k_proj<false><<<gp, blk, 0, stream>>>(query, wQt, qws);
    k_proj<false><<<gp, blk, 0, stream>>>(key,   wKt, kws);
    k_proj<true ><<<gp, blk, 0, stream>>>(value, wVt, vTw);        // store V^T

    k_attn<<<dim3(Ln / 16, Hn, Bn), blk, 0, stream>>>(qws, kws, vTw, mask, aw, out_p, xh);

    k_fc<<<dim3((Bn * Ln) / 16, Dn / 64), blk, 0, stream>>>(xh, wFt, out_x);
}
